// MultiSizePatchEmbed_68564857914001
// MI455X (gfx1250) — matrix operand order repaired and checked by reading, not compiled, not run on hardware
//
#include <hip/hip_runtime.h>
#include <math.h>

typedef __bf16 bf16;
typedef __attribute__((ext_vector_type(16))) __bf16 v16bf;
typedef __attribute__((ext_vector_type(8)))  __bf16 v8bf;
typedef __attribute__((ext_vector_type(8)))  float  v8f;

#define EMBED 1024
#define INTER 4096
#define NCOMB 5120   // INTER + EMBED
#define NTOK  512
#define MROWS 8192   // 16 layers * 512 tokens per group
#define MID_M 24576  // 48 * 512

// ---------------- CDNA5 async-copy helpers ----------------

// GLOBAL_LOAD_ASYNC_TO_LDS_B128: per-lane 16B  global -> LDS, tracked by ASYNCcnt.
// LDS operand is the wave-relative LDS byte offset (= low 32 bits of the flat
// pointer, since local->flat cast is {aperture_hi32, lds_offset}).
__device__ __forceinline__ void async_copy_b128(const bf16* gptr, bf16* lptr) {
    unsigned lds_off = (unsigned)(uintptr_t)lptr;
    unsigned long long gaddr = (unsigned long long)(uintptr_t)gptr;
    asm volatile("global_load_async_to_lds_b128 %0, %1, off"
                 :: "v"(lds_off), "v"(gaddr) : "memory");
}

__device__ __forceinline__ void wait_async0() {
    asm volatile("s_wait_asynccnt 0x0" ::: "memory");
}

// ---------------- prep kernels ----------------

__global__ void f32_to_bf16_kernel(const float* __restrict__ in, bf16* __restrict__ out, int n) {
    int i = blockIdx.x * blockDim.x + threadIdx.x;
    if (i < n) out[i] = (bf16)in[i];
}

// xs = concat([x,m,t], -1), zero-padded to Kp, bf16. Layout [8192, Kp].
__global__ void pack_xs_kernel(const float* __restrict__ x, const float* __restrict__ m,
                               const float* __restrict__ t, bf16* __restrict__ out,
                               int p, int Kp) {
    int i = blockIdx.x * blockDim.x + threadIdx.x;
    int total = MROWS * Kp;
    if (i >= total) return;
    int r = i / Kp;
    int k = i - r * Kp;
    float v = 0.0f;
    if (k < p)            v = x[r * p + k];
    else if (k < 2 * p)   v = m[r * p + (k - p)];
    else if (k < 3 * p)   v = t[r * p + (k - 2 * p)];
    out[i] = (bf16)v;
}

// Combined interpolated weight: rows [0,4096) from W_lin, [4096,5120) from W_res.
// F.interpolate(mode='linear', align_corners=False) along the 96-wide axis -> p3,
// zero-padded to Kp. Layout [5120, Kp] (row-major, K contiguous => B as [N][K]).
__global__ void interp_w_kernel(const float* __restrict__ Wl, const float* __restrict__ Wr,
                                bf16* __restrict__ out, int p3, int Kp) {
    int i = blockIdx.x * blockDim.x + threadIdx.x;
    int total = NCOMB * Kp;
    if (i >= total) return;
    int row = i / Kp;
    int k = i - row * Kp;
    float v = 0.0f;
    if (k < p3) {
        float scale = 96.0f / (float)p3;
        float src = ((float)k + 0.5f) * scale - 0.5f;
        src = fminf(fmaxf(src, 0.0f), 95.0f);
        int i0 = (int)src;
        int i1 = (i0 + 1 < 95) ? (i0 + 1) : 95;
        float tt = src - (float)i0;
        const float* Wrow = (row < INTER) ? (Wl + (size_t)row * 96)
                                          : (Wr + (size_t)(row - INTER) * 96);
        v = Wrow[i0] * (1.0f - tt) + Wrow[i1] * tt;
    }
    out[i] = (bf16)v;
}

// ---------------- WMMA GEMM, double-buffered async pipeline ----------------
// C[M,N] = A[M,K] * B[N,K]^T   (A row-major, B stored [N][K] K-contiguous)
// Block tile 128x128, 256 threads (8 waves as 2x4), wave tile 64x32 (4x2 wmma),
// K-step 32. Global->LDS via GLOBAL_LOAD_ASYNC_TO_LDS_B128 (ASYNCcnt), two LDS
// buffers so tile i+1 copies overlap tile i WMMAs; one barrier per K-step.
// EPI=0: layer-2: out[m*1024+n] += acc
// EPI=1: layer-1: n<4096 -> mid[dst*4096+n] = bf16(silu(acc+b_lin[n]))
//                 n>=4096 -> out[dst*1024+(n-4096)] = acc + b_res + b_out
template <int EPI>
__global__ __launch_bounds__(256)
void wmma_gemm_kernel(const bf16* __restrict__ A, const bf16* __restrict__ B, int K,
                      const int* __restrict__ idx,
                      const float* __restrict__ b_lin, const float* __restrict__ b_res,
                      const float* __restrict__ b_out,
                      bf16* __restrict__ mid, float* __restrict__ out) {
    __shared__ __align__(32) bf16 lds_a[2][128][32];
    __shared__ __align__(32) bf16 lds_b[2][128][32];

    const int tid  = threadIdx.x;
    const int lane = tid & 31;
    const int wave = tid >> 5;
    const int wm = wave >> 2;        // 0..1
    const int wn = wave & 3;         // 0..3
    const int bm = blockIdx.y * 128;
    const int bn = blockIdx.x * 128;

    const int lr = lane & 15;        // row/col within 16
    const int kh = lane >> 4;        // K-half select

    const int r0 = tid >> 2;          // 0..63
    const int c8 = (tid & 3) << 3;    // 0,8,16,24

    v8f acc[4][2];
#pragma unroll
    for (int mi = 0; mi < 4; ++mi)
#pragma unroll
        for (int ni = 0; ni < 2; ++ni)
            acc[mi][ni] = (v8f){0.f, 0.f, 0.f, 0.f, 0.f, 0.f, 0.f, 0.f};

    // 4 async b128 copies per thread per tile (2 rows of A, 2 rows of B)
    auto issue_tile = [&](int buf, int kk) {
#pragma unroll
        for (int it = 0; it < 2; ++it) {
            int r = r0 + it * 64;
            async_copy_b128(&A[(size_t)(bm + r) * K + kk + c8], &lds_a[buf][r][c8]);
            async_copy_b128(&B[(size_t)(bn + r) * K + kk + c8], &lds_b[buf][r][c8]);
        }
    };

    const int nk = K >> 5;
    issue_tile(0, 0);

    for (int i = 0; i < nk; ++i) {
        wait_async0();        // this wave's tile-i copies have landed in LDS
        __syncthreads();      // everyone's landed; everyone done reading buf we write next
        if (i + 1 < nk) issue_tile((i + 1) & 1, (i + 1) << 5);  // overlaps compute below
        const int cb = i & 1;

        // B fragments: lane<16 holds col N=lr with K 0..15; lane>=16 holds K 16..31.
        v16bf b_frag[2];
#pragma unroll
        for (int ni = 0; ni < 2; ++ni) {
            int bc = wn * 32 + ni * 16 + lr;
            b_frag[ni] = *(const v16bf*)&lds_b[cb][bc][kh * 16];
        }

#pragma unroll
        for (int mi = 0; mi < 4; ++mi) {
            // A fragment: lane<16 row M=lr, elts 0-15 = K0-15;
            //             lane>=16 same row, K16-31.
            int am = wm * 64 + mi * 16 + lr;
            v16bf a_frag = *(const v16bf*)&lds_a[cb][am][kh * 16];
#pragma unroll
            for (int ni = 0; ni < 2; ++ni) {
                acc[mi][ni] = __builtin_amdgcn_wmma_f32_16x16x32_bf16(
                    false, a_frag, false, b_frag[ni], (short)0, acc[mi][ni], false, false);
            }
        }
    }

    // Epilogue. C/D layout: N = lane&15; lanes 0-15 hold M = v, lanes 16-31 M = 8+v.
#pragma unroll
    for (int mi = 0; mi < 4; ++mi) {
#pragma unroll
        for (int ni = 0; ni < 2; ++ni) {
#pragma unroll
            for (int v = 0; v < 8; ++v) {
                int m = bm + wm * 64 + mi * 16 + kh * 8 + v;
                int n = bn + wn * 32 + ni * 16 + lr;
                float val = acc[mi][ni][v];
                if (EPI == 0) {
                    out[(size_t)m * EMBED + n] += val;
                } else {
                    int gl = m >> 9;           // layer within group
                    int nn = m & 511;          // token
                    size_t dst = (size_t)idx[gl] * NTOK + nn;
                    if (n < INTER) {
                        float y = val + b_lin[n];
                        float s = y / (1.0f + __expf(-y));   // silu
                        mid[dst * INTER + n] = (bf16)s;
                    } else {
                        int h = n - INTER;
                        out[dst * EMBED + h] = val + b_res[h] + b_out[h];
                    }
                }
            }
        }
    }
}

// ---------------- host launcher ----------------

extern "C" void kernel_launch(void* const* d_in, const int* in_sizes, int n_in,
                              void* d_out, int out_size, void* d_ws, size_t ws_size,
                              hipStream_t stream) {
    const float* X[3] = {(const float*)d_in[0], (const float*)d_in[3], (const float*)d_in[6]};
    const float* M[3] = {(const float*)d_in[1], (const float*)d_in[4], (const float*)d_in[7]};
    const float* T[3] = {(const float*)d_in[2], (const float*)d_in[5], (const float*)d_in[8]};
    const int*   IDX[3] = {(const int*)d_in[9], (const int*)d_in[10], (const int*)d_in[11]};
    const float* W_lin = (const float*)d_in[12];
    const float* b_lin = (const float*)d_in[13];
    const float* W_res = (const float*)d_in[14];
    const float* b_res = (const float*)d_in[15];
    const float* W_out = (const float*)d_in[16];
    const float* b_out = (const float*)d_in[17];
    float* out = (float*)d_out;

    const int P[3]  = {16, 32, 64};
    const int KP[3] = {64, 96, 192};   // 3p padded to multiple of 32

    // workspace carve-out (deterministic)
    char* ws = (char*)d_ws;
    size_t off = 0;
    auto wsAlloc = [&](size_t bytes) -> void* {
        void* p = (void*)(ws + off);
        off += (bytes + 255) & ~(size_t)255;
        return p;
    };
    bf16* Wout_bf = (bf16*)wsAlloc((size_t)EMBED * INTER * sizeof(bf16));
    bf16* mid     = (bf16*)wsAlloc((size_t)MID_M * INTER * sizeof(bf16));
    bf16* xs[3];
    bf16* wc[3];
    for (int g = 0; g < 3; ++g) {
        xs[g] = (bf16*)wsAlloc((size_t)MROWS * KP[g] * sizeof(bf16));
        wc[g] = (bf16*)wsAlloc((size_t)NCOMB * KP[g] * sizeof(bf16));
    }

    // 1) W_out -> bf16
    {
        int n = EMBED * INTER;
        f32_to_bf16_kernel<<<(n + 255) / 256, 256, 0, stream>>>(W_out, Wout_bf, n);
    }
    // 2) pack xs + interp weights per group
    for (int g = 0; g < 3; ++g) {
        int tot_x = MROWS * KP[g];
        pack_xs_kernel<<<(tot_x + 255) / 256, 256, 0, stream>>>(X[g], M[g], T[g], xs[g],
                                                                P[g], KP[g]);
        int tot_w = NCOMB * KP[g];
        interp_w_kernel<<<(tot_w + 255) / 256, 256, 0, stream>>>(W_lin, W_res, wc[g],
                                                                 3 * P[g], KP[g]);
    }
    // 3) layer-1 GEMMs: [8192 x Kp] x [5120 x Kp]^T, scatter epilogue
    for (int g = 0; g < 3; ++g) {
        dim3 grid(NCOMB / 128, MROWS / 128);  // 40 x 64
        wmma_gemm_kernel<1><<<grid, 256, 0, stream>>>(xs[g], wc[g], KP[g], IDX[g],
                                                      b_lin, b_res, b_out, mid, out);
    }
    // 4) layer-2 GEMM: [24576 x 4096] x [1024 x 4096]^T, accumulate into out
    {
        dim3 grid(EMBED / 128, MID_M / 128);  // 8 x 192
        wmma_gemm_kernel<0><<<grid, 256, 0, stream>>>(mid, Wout_bf, INTER, nullptr,
                                                      nullptr, nullptr, nullptr,
                                                      nullptr, out);
    }
}
